// ResidualMamba_43258910605793
// MI455X (gfx1250) — matrix operand order repaired and checked by reading, not compiled, not run on hardware
//
#include <hip/hip_runtime.h>
#include <hip/hip_bf16.h>
#include <math.h>

// ---------------------------------------------------------------------------
// Mamba block for MI455X (gfx1250, wave32).
// GEMMs: TDM (tensor_load_to_lds) double-buffered LDS staging +
//        v_wmma_f32_16x16x32_bf16 (bf16 in, f32 accumulate).
// Selective scan: lane==state (32 lanes == D_STATE), 4 channels per wave.
// ---------------------------------------------------------------------------

#define B_    4
#define L_    2048
#define DM_   1024        // d_model
#define DI_   2048        // d_inner
#define DTR_  64          // dt_rank
#define DS_   32          // d_state
#define BL_   (B_ * L_)   // 8192 rows

typedef __attribute__((ext_vector_type(16))) __bf16 v16bf;
typedef __attribute__((ext_vector_type(8)))  __bf16 v8bf;
typedef __attribute__((ext_vector_type(8)))  float  v8f;
typedef __attribute__((ext_vector_type(4)))  unsigned int v4u;
typedef __attribute__((ext_vector_type(8)))  int v8i;
typedef __attribute__((ext_vector_type(4)))  int v4i;

static __device__ __forceinline__ float sigmoidf_(float x) {
    return 1.f / (1.f + __expf(-x));
}
static __device__ __forceinline__ float softplusf_(float x) {
    return (x > 20.f) ? x : __logf(1.f + __expf(x));
}

// ---------------------------------------------------------------------------
// TDM: 2D tile load (bf16 elements) Global -> LDS via Tensor Data Mover.
// Tile = tile_k (contiguous K elems) x tile_rows, row stride = ld elems.
// Descriptor packing per CDNA5 ISA D# group0/group1 layout.
// ---------------------------------------------------------------------------
static __device__ __forceinline__ void tdm_load_tile_bf16(
    unsigned lds_off, const __bf16* gptr,
    unsigned tile_k, unsigned tile_rows,
    unsigned k_total, unsigned rows_total, unsigned ld)
{
    const unsigned long long ga = (unsigned long long)(uintptr_t)gptr;
    v4u g0;
    g0[0] = 1u;                                   // count=1, user descriptor
    g0[1] = lds_off;                              // lds_addr (bytes)
    g0[2] = (unsigned)(ga & 0xffffffffull);       // global_addr[31:0]
    g0[3] = (unsigned)((ga >> 32) & 0x1ffffffull) // global_addr[56:32]
          | (2u << 30);                           // type = 2 ("image")
    v8i g1;
    g1[0] = (int)(1u << 16);                      // workgroup_mask=0, data_size=1 (2B)
    g1[1] = (int)((k_total & 0xffffu) << 16);     // tensor_dim0[15:0]
    g1[2] = (int)((k_total >> 16) | ((rows_total & 0xffffu) << 16)); // dim0 hi, dim1 lo
    g1[3] = (int)((rows_total >> 16) | (tile_k << 16));              // dim1 hi, tile_dim0
    g1[4] = (int)(tile_rows & 0xffffu);           // tile_dim1 (tile_dim2 = 0)
    g1[5] = (int)ld;                              // tensor_dim0_stride[31:0]
    g1[6] = 0;                                    // stride hi / dim1_stride lo
    g1[7] = 0;
    v4i zz = (v4i)0;
#if defined(__clang_major__) && (__clang_major__ >= 23)
    __builtin_amdgcn_tensor_load_to_lds(g0, g1, zz, zz, (v8i)0, 0);
#else
    __builtin_amdgcn_tensor_load_to_lds(g0, g1, zz, zz, 0);
#endif
}

// ---------------------------------------------------------------------------
// LayerNorm (row = 1024 cols) -> bf16 output
// ---------------------------------------------------------------------------
__global__ __launch_bounds__(256)
void ln_kernel(const float* __restrict__ x, const float* __restrict__ g,
               const float* __restrict__ b, __bf16* __restrict__ out)
{
    const int row = blockIdx.x;
    const float* xr = x + (size_t)row * DM_;
    float s = 0.f, ss = 0.f;
    for (int i = threadIdx.x; i < DM_; i += 256) {
        float v = xr[i]; s += v; ss += v * v;
    }
    for (int off = 16; off; off >>= 1) {
        s  += __shfl_xor(s,  off);
        ss += __shfl_xor(ss, off);
    }
    __shared__ float sh0[8], sh1[8];
    if ((threadIdx.x & 31) == 0) { sh0[threadIdx.x >> 5] = s; sh1[threadIdx.x >> 5] = ss; }
    __syncthreads();
    s = 0.f; ss = 0.f;
    #pragma unroll
    for (int w = 0; w < 8; ++w) { s += sh0[w]; ss += sh1[w]; }
    const float mu  = s * (1.f / DM_);
    const float var = ss * (1.f / DM_) - mu * mu;
    const float r   = rsqrtf(var + 1e-5f);
    for (int i = threadIdx.x; i < DM_; i += 256) {
        float v = (xr[i] - mu) * r * g[i] + b[i];
        out[(size_t)row * DM_ + i] = (__bf16)v;
    }
}

// ---------------------------------------------------------------------------
// f32 -> bf16 casts (flat and strided-submatrix)
// ---------------------------------------------------------------------------
__global__ __launch_bounds__(256)
void cast_kernel(const float* __restrict__ src, __bf16* __restrict__ dst, size_t n)
{
    size_t i = (size_t)blockIdx.x * 256 + threadIdx.x;
    if (i < n) dst[i] = (__bf16)src[i];
}

__global__ __launch_bounds__(256)
void cast_sub_kernel(const float* __restrict__ src, __bf16* __restrict__ dst,
                     int rows, int cols, int src_ld)
{
    size_t i = (size_t)blockIdx.x * 256 + threadIdx.x;
    size_t n = (size_t)rows * cols;
    if (i >= n) return;
    int r = (int)(i / cols), c = (int)(i % cols);
    dst[i] = (__bf16)src[(size_t)r * src_ld + c];
}

// ---------------------------------------------------------------------------
// bf16 WMMA GEMM with TDM double-buffered LDS staging.
// C[m,n] = sum_k A[m,k] * Bw[n,k]  (Bw row-major (N,K), einsum 'bld,ed->ble').
// Block: 8 waves, block tile 128Mx64N; wave tile 16Mx64N (4 accumulators).
// Per K-step (32): TDM loads A-tile (128x32) and B-tile (64x32) into LDS;
// wave 0 issues tile i+1 while all waves compute tile i from LDS.
// Epilogues: 0 = store, 1 = +extra[m*ldc+n], 2 = softplus(acc + extra[n]).
// ---------------------------------------------------------------------------
__global__ __launch_bounds__(256)
void gemm_bf16_wmma(const __bf16* __restrict__ A, const __bf16* __restrict__ Bw,
                    float* __restrict__ C, int M, int N, int K,
                    int lda, int ldb, int ldc,
                    const float* __restrict__ extra, int epi)
{
    __shared__ __bf16 shA[2][128 * 32];   // [buf][row*32 + k]
    __shared__ __bf16 shB[2][64 * 32];    // [buf][row*32 + k]

    const int lane = threadIdx.x & 31;
    const int wave = threadIdx.x >> 5;
    const int m0 = blockIdx.y * 128 + wave * 16;
    const int n0 = blockIdx.x * 64;
    if (m0 >= M || n0 >= N) return;

    const int hi = lane >> 4;   // K half selector per WMMA operand layout
    const int mr = lane & 15;   // A row within 16 / B column within 16

    const unsigned ldsA0 = (unsigned)(uintptr_t)(void*)&shA[0][0];
    const unsigned ldsA1 = (unsigned)(uintptr_t)(void*)&shA[1][0];
    const unsigned ldsB0 = (unsigned)(uintptr_t)(void*)&shB[0][0];
    const unsigned ldsB1 = (unsigned)(uintptr_t)(void*)&shB[1][0];

    const __bf16* agbase = A  + (size_t)(blockIdx.y * 128) * lda;
    const __bf16* bgbase = Bw + (size_t)n0 * ldb;

    const int nk = K / 32;

    // Prologue: tile 0 -> buffer 0
    if (threadIdx.x < 32) {
        tdm_load_tile_bf16(ldsA0, agbase, 32, 128, K, M, lda);
        tdm_load_tile_bf16(ldsB0, bgbase, 32, 64,  K, N, ldb);
    }

    v8f acc0 = (v8f)0.f, acc1 = (v8f)0.f, acc2 = (v8f)0.f, acc3 = (v8f)0.f;

    for (int i = 0; i < nk; ++i) {
        const int cur = i & 1;
        if (threadIdx.x < 32) {
            if (i + 1 < nk) {
                const int k1 = (i + 1) * 32;
                tdm_load_tile_bf16(cur ? ldsA0 : ldsA1, agbase + k1, 32, 128, K, M, lda);
                tdm_load_tile_bf16(cur ? ldsB0 : ldsB1, bgbase + k1, 32, 64,  K, N, ldb);
                __builtin_amdgcn_s_wait_tensorcnt(2);  // in-order TDM: tile i done
            } else {
                __builtin_amdgcn_s_wait_tensorcnt(0);
            }
        }
        __syncthreads();   // tile i visible to all waves

        // A fragment (16-bit A 16x32 layout, contiguous K to match B):
        //  lanes 0-15: elems 0-15 -> K=0..15
        //  lanes16-31: elems 0-15 -> K=16..31
        const __bf16* arow = &shA[cur][(wave * 16 + mr) * 32];
        v8bf alo = *(const v8bf*)(arow + hi * 16);
        v8bf ahi = *(const v8bf*)(arow + hi * 16 + 8);
        v16bf afrag = __builtin_shufflevector(alo, ahi,
            0,1,2,3,4,5,6,7,8,9,10,11,12,13,14,15);

        // B fragment (16-bit B 32x16): lane holds column n=mr, 16 consecutive
        // K values; lanes 16-31 hold the upper K half.
        v16bf bf0 = *(const v16bf*)(&shB[cur][( 0 + mr) * 32 + hi * 16]);
        v16bf bf1 = *(const v16bf*)(&shB[cur][(16 + mr) * 32 + hi * 16]);
        v16bf bf2 = *(const v16bf*)(&shB[cur][(32 + mr) * 32 + hi * 16]);
        v16bf bf3 = *(const v16bf*)(&shB[cur][(48 + mr) * 32 + hi * 16]);

        acc0 = __builtin_amdgcn_wmma_f32_16x16x32_bf16(false, afrag, false, bf0,
                                                       (short)0, acc0, false, false);
        acc1 = __builtin_amdgcn_wmma_f32_16x16x32_bf16(false, afrag, false, bf1,
                                                       (short)0, acc1, false, false);
        acc2 = __builtin_amdgcn_wmma_f32_16x16x32_bf16(false, afrag, false, bf2,
                                                       (short)0, acc2, false, false);
        acc3 = __builtin_amdgcn_wmma_f32_16x16x32_bf16(false, afrag, false, bf3,
                                                       (short)0, acc3, false, false);

        __syncthreads();   // all reads of buf[cur] done before it is re-filled
    }

    // C/D layout: VGPR r -> (lanes 0-15: M=m0+r), (lanes16-31: M=m0+8+r), N=n0+lane%16
    v8f accs[4] = {acc0, acc1, acc2, acc3};
    #pragma unroll
    for (int j = 0; j < 4; ++j) {
        #pragma unroll
        for (int r = 0; r < 8; ++r) {
            const int m = m0 + r + hi * 8;
            const int n = n0 + j * 16 + mr;
            float v = accs[j][r];
            const size_t idx = (size_t)m * ldc + n;
            if (epi == 1)       v += extra[idx];
            else if (epi == 2)  v = softplusf_(v + extra[n]);
            C[idx] = v;
        }
    }
}

// ---------------------------------------------------------------------------
// Causal depthwise conv (K=4) + bias + SiLU -> u (f32) and u (bf16)
// ---------------------------------------------------------------------------
__global__ __launch_bounds__(256)
void conv_silu_kernel(const float* __restrict__ xz, const float* __restrict__ w,
                      const float* __restrict__ bias,
                      float* __restrict__ u, __bf16* __restrict__ ubf)
{
    const size_t idx = (size_t)blockIdx.x * 256 + threadIdx.x; // over BL_*DI_
    const int d = (int)(idx % DI_);
    const size_t bl = idx / DI_;
    const int l = (int)(bl % L_);
    const size_t brow = bl - l;     // b*L
    float acc = bias[d];
    #pragma unroll
    for (int j = 0; j < 4; ++j) {
        const int lp = l - 3 + j;
        if (lp >= 0)
            acc += xz[(brow + lp) * (size_t)(2 * DI_) + d] * w[d * 4 + j];
    }
    const float uval = acc * sigmoidf_(acc);
    u[idx]   = uval;
    ubf[idx] = (__bf16)uval;
}

// ---------------------------------------------------------------------------
// Selective scan.  lane == state n (wave32 == D_STATE), 4 channels per wave.
// Fuses +u*D, *silu(z) gating, bf16 store for out_proj.
// ---------------------------------------------------------------------------
__global__ __launch_bounds__(256)
void scan_kernel(const float* __restrict__ xdbl,   // (BL_, 128): [dt|B|C]
                 const float* __restrict__ delta,  // (BL_, DI_)
                 const float* __restrict__ u,      // (BL_, DI_)
                 const float* __restrict__ xz,     // (BL_, 2*DI_): z at +DI_
                 const float* __restrict__ A_log,  // (DI_, DS_)
                 const float* __restrict__ Dp,     // (DI_)
                 __bf16* __restrict__ ybf)         // (BL_, DI_)
{
    const int lane = threadIdx.x & 31;   // state index n
    const int wave = threadIdx.x >> 5;   // 0..7
    const int gid  = blockIdx.x;         // 0..255
    const int b    = gid >> 6;           // 64 channel-groups of 32 per batch
    const int d0   = (gid & 63) * 32 + wave * 4;

    float An[4], Dc[4], h[4];
    #pragma unroll
    for (int c = 0; c < 4; ++c) {
        An[c] = -__expf(A_log[(size_t)(d0 + c) * DS_ + lane]);
        Dc[c] = Dp[d0 + c];
        h[c]  = 0.f;
    }

    for (int t = 0; t < L_; ++t) {
        const size_t row = (size_t)b * L_ + t;
        const float Bn = xdbl[row * 128 + DTR_ + lane];        // B(t)[n]
        const float Cn = xdbl[row * 128 + DTR_ + DS_ + lane];  // C(t)[n]
        #pragma unroll
        for (int c = 0; c < 4; ++c) {
            const float dl = delta[row * DI_ + d0 + c];
            const float ut = u[row * DI_ + d0 + c];
            const float dA = __expf(dl * An[c]);
            h[c] = dA * h[c] + (dl * ut) * Bn;
            float p = h[c] * Cn;
            #pragma unroll
            for (int off = 16; off; off >>= 1) p += __shfl_xor(p, off);
            const float zt = xz[row * (size_t)(2 * DI_) + DI_ + d0 + c];
            float y = p + ut * Dc[c];
            y *= zt * sigmoidf_(zt);
            if (lane == c) ybf[row * DI_ + d0 + c] = (__bf16)y;
        }
    }
}

// ---------------------------------------------------------------------------
// Workspace layout (bytes, all 256-aligned)
// ---------------------------------------------------------------------------
#define OFF_XN     ((size_t)0)                                  // bf16 BL*DM
#define OFF_WIN    (OFF_XN    + (size_t)BL_*DM_*2)              // bf16 2DI*DM
#define OFF_WXP    (OFF_WIN   + (size_t)2*DI_*DM_*2)            // bf16 128*DI
#define OFF_WDT    (OFF_WXP   + (size_t)(DTR_+2*DS_)*DI_*2)     // bf16 DI*DTR
#define OFF_WOUT   (OFF_WDT   + (size_t)DI_*DTR_*2)             // bf16 DM*DI
#define OFF_XZ     (OFF_WOUT  + (size_t)DM_*DI_*2)              // f32 BL*2DI
#define OFF_U      (OFF_XZ    + (size_t)BL_*2*DI_*4)            // f32 BL*DI
#define OFF_UBF    (OFF_U     + (size_t)BL_*DI_*4)              // bf16 BL*DI
#define OFF_XDBL   (OFF_UBF   + (size_t)BL_*DI_*2)              // f32 BL*128
#define OFF_DTBF   (OFF_XDBL  + (size_t)BL_*128*4)              // bf16 BL*64
#define OFF_DELTA  (OFF_DTBF  + (size_t)BL_*DTR_*2)             // f32 BL*DI
#define OFF_YBF    (OFF_DELTA + (size_t)BL_*DI_*4)              // bf16 BL*DI

extern "C" void kernel_launch(void* const* d_in, const int* in_sizes, int n_in,
                              void* d_out, int out_size, void* d_ws, size_t ws_size,
                              hipStream_t stream)
{
    const float* x        = (const float*)d_in[0];
    const float* ln_g     = (const float*)d_in[1];
    const float* ln_b     = (const float*)d_in[2];
    const float* in_w     = (const float*)d_in[3];   // (4096,1024)
    const float* conv_w   = (const float*)d_in[4];   // (2048,4)
    const float* conv_b   = (const float*)d_in[5];
    const float* xp_w     = (const float*)d_in[6];   // (128,2048)
    const float* dt_w     = (const float*)d_in[7];   // (2048,64)
    const float* dt_b     = (const float*)d_in[8];
    const float* A_log    = (const float*)d_in[9];   // (2048,32)
    const float* D_param  = (const float*)d_in[10];
    const float* out_w    = (const float*)d_in[11];  // (1024,2048)
    float* out            = (float*)d_out;

    char* ws = (char*)d_ws;
    __bf16* xn_bf   = (__bf16*)(ws + OFF_XN);
    __bf16* win_bf  = (__bf16*)(ws + OFF_WIN);
    __bf16* wxp_bf  = (__bf16*)(ws + OFF_WXP);
    __bf16* wdt_bf  = (__bf16*)(ws + OFF_WDT);
    __bf16* wout_bf = (__bf16*)(ws + OFF_WOUT);
    float*  xz      = (float*)(ws + OFF_XZ);
    float*  u       = (float*)(ws + OFF_U);
    __bf16* u_bf    = (__bf16*)(ws + OFF_UBF);
    float*  xdbl    = (float*)(ws + OFF_XDBL);
    __bf16* dt_bf   = (__bf16*)(ws + OFF_DTBF);
    float*  delta   = (float*)(ws + OFF_DELTA);
    __bf16* y_bf    = (__bf16*)(ws + OFF_YBF);

    // 1) weight casts to bf16
    {
        size_t n;
        n = (size_t)2*DI_*DM_;
        cast_kernel<<<(n + 255) / 256, 256, 0, stream>>>(in_w, win_bf, n);
        n = (size_t)(DTR_ + 2*DS_) * DI_;
        cast_kernel<<<(n + 255) / 256, 256, 0, stream>>>(xp_w, wxp_bf, n);
        n = (size_t)DI_*DTR_;
        cast_kernel<<<(n + 255) / 256, 256, 0, stream>>>(dt_w, wdt_bf, n);
        n = (size_t)DM_*DI_;
        cast_kernel<<<(n + 255) / 256, 256, 0, stream>>>(out_w, wout_bf, n);
    }

    // 2) LayerNorm -> xn (bf16)
    ln_kernel<<<BL_, 256, 0, stream>>>(x, ln_g, ln_b, xn_bf);

    // 3) in_proj GEMM: (8192,1024) x (4096,1024)^T -> xz (8192,4096)
    gemm_bf16_wmma<<<dim3((2*DI_)/64, BL_/128), 256, 0, stream>>>(
        xn_bf, win_bf, xz, BL_, 2*DI_, DM_, DM_, DM_, 2*DI_, nullptr, 0);

    // 4) causal depthwise conv + SiLU -> u
    conv_silu_kernel<<<((size_t)BL_*DI_)/256, 256, 0, stream>>>(
        xz, conv_w, conv_b, u, u_bf);

    // 5) x_proj GEMM: (8192,2048) x (128,2048)^T -> x_dbl (8192,128)
    gemm_bf16_wmma<<<dim3(128/64, BL_/128), 256, 0, stream>>>(
        u_bf, wxp_bf, xdbl, BL_, DTR_ + 2*DS_, DI_, DI_, DI_, DTR_ + 2*DS_,
        nullptr, 0);

    // 6) dt slice cast (cols 0..63 of x_dbl) -> bf16
    {
        size_t n = (size_t)BL_ * DTR_;
        cast_sub_kernel<<<(n + 255) / 256, 256, 0, stream>>>(
            xdbl, dt_bf, BL_, DTR_, DTR_ + 2*DS_);
    }

    // 7) delta GEMM: (8192,64) x (2048,64)^T + bias -> softplus -> delta
    gemm_bf16_wmma<<<dim3(DI_/64, BL_/128), 256, 0, stream>>>(
        dt_bf, wdt_bf, delta, BL_, DI_, DTR_, DTR_, DTR_, DI_, dt_b, 2);

    // 8) selective scan (+u*D, *silu(z)) -> y (bf16)
    scan_kernel<<<256, 256, 0, stream>>>(xdbl, delta, u, xz, A_log, D_param, y_bf);

    // 9) out_proj GEMM: (8192,2048) x (1024,2048)^T + residual -> out
    gemm_bf16_wmma<<<dim3(DM_/64, BL_/128), 256, 0, stream>>>(
        y_bf, wout_bf, out, BL_, DM_, DI_, DI_, DI_, DM_, x, 1);
}
